// RelationalMemory_86749749445273
// MI455X (gfx1250) — matrix operand order repaired and checked by reading, not compiled, not run on hardware
//
#include <hip/hip_runtime.h>
#include <hip/hip_bf16.h>

// ---------------------------------------------------------------------------
// RelationalMemory forward for MI455X (gfx1250), wave32 + WMMA bf16 + TDM.
// ---------------------------------------------------------------------------

typedef __attribute__((ext_vector_type(16))) __bf16        v16bf;
typedef __attribute__((ext_vector_type(8)))  float         v8f;
typedef __attribute__((ext_vector_type(8)))  unsigned int  v8u;
typedef __attribute__((ext_vector_type(4)))  int           v4i;
typedef __attribute__((ext_vector_type(8)))  int           v8i;

#define BB   2048
#define TT   64
#define MM   256
#define SS   8
#define GG   512
#define BT   (BB * TT)        // 131072
#define BSM  (BB * SS * MM)   // 4194304

__device__ __forceinline__ float bf2f(unsigned short h) {
  union { unsigned int u; float f; } v;
  v.u = ((unsigned int)h) << 16;
  return v.f;
}
__device__ __forceinline__ unsigned short f2bf(float f) {
  union { float f; unsigned int u; } v;
  v.f = f;
  unsigned int r = v.u + 0x7fffu + ((v.u >> 16) & 1u);
  return (unsigned short)(r >> 16);
}
__device__ __forceinline__ unsigned int pack2(float a, float b) {
  return (unsigned int)f2bf(a) | ((unsigned int)f2bf(b) << 16);
}
__device__ __forceinline__ float sigmoidf(float x) {
  return 1.0f / (1.0f + __expf(-x));
}

// LDS byte address (32-bit, relative to workgroup LDS base) of a shared ptr.
typedef __attribute__((address_space(3))) unsigned int lds_u32_t;
__device__ __forceinline__ unsigned lds_addr_of(unsigned int* p) {
  return (unsigned)(unsigned long long)(lds_u32_t*)p;
}

// ---------------------------------------------------------------------------
// Generic tiled GEMM: C[M,N] = act(A[M,K] @ W[K,N] + bias), W given as
// Bt[N,K] bf16 (pre-transposed). Block tile 128x128, 8 waves (2M x 4N),
// wave tile 64x32 = 4x2 WMMA 16x16x32 bf16 tiles, fp32 accumulate.
// Bt tile (128 rows x 64B, +16B LDS pad per row) is fetched by the Tensor
// Data Mover (tensor_load_to_lds + s_wait_tensorcnt), issued by wave 0.
// ---------------------------------------------------------------------------
template <bool A_BF16, bool OUT_BF16, bool RELU>
__global__ __launch_bounds__(256) void gemm_wmma_kernel(
    const void* __restrict__ Ap, const unsigned short* __restrict__ Bt,
    const float* __restrict__ bias, void* __restrict__ Cp,
    int Mrows, int Ncols, int Kdim) {
  constexpr int LSTR = 20;  // u32 per LDS row: 16 used + 4 pad (80B, 16B-aligned)
  __shared__ unsigned int ldsA[128 * LSTR];
  __shared__ unsigned int ldsB[128 * LSTR];

  const int tid   = threadIdx.x;
  const int lane  = tid & 31;
  const int wid   = tid >> 5;
  const int waveM = wid >> 2;   // 0..1
  const int waveN = wid & 3;    // 0..3
  const int tileM0 = blockIdx.y * 128;
  const int tileN0 = blockIdx.x * 128;

  v8f acc[4][2];
  v8f zero = {0.f, 0.f, 0.f, 0.f, 0.f, 0.f, 0.f, 0.f};
#pragma unroll
  for (int i = 0; i < 4; ++i)
#pragma unroll
    for (int j = 0; j < 2; ++j) acc[i][j] = zero;

  const int loadRow = tid >> 1;          // 0..127
  const int loadOff = (tid & 1) * 16;    // element offset within K-32 slab
  unsigned int* aDst = &ldsA[loadRow * LSTR + (loadOff >> 1)];

  const int row16   = lane & 15;   // M (A frag) / N (B frag) within 16
  const int halfSel = lane >> 4;   // lane half selects K sub-range

  const unsigned ldsB_base = lds_addr_of(&ldsB[0]);

  for (int kb = 0; kb < Kdim; kb += 32) {
    // ---- stage A tile (128 x 32 bf16) via per-thread path ----
    if (A_BF16) {
      const unsigned short* A = (const unsigned short*)Ap;
      const uint4* src =
          (const uint4*)(A + (size_t)(tileM0 + loadRow) * Kdim + kb + loadOff);
      uint4 p0 = src[0], p1 = src[1];
      *(uint4*)(aDst)     = p0;
      *(uint4*)(aDst + 4) = p1;
      if (kb + 32 < Kdim) __builtin_prefetch((const char*)(src) + 64, 0, 1);
    } else {
      const float* A = (const float*)Ap;
      const float* src = A + (size_t)(tileM0 + loadRow) * Kdim + kb + loadOff;
      float4 f0 = *(const float4*)(src + 0);
      float4 f1 = *(const float4*)(src + 4);
      float4 f2 = *(const float4*)(src + 8);
      float4 f3 = *(const float4*)(src + 12);
      aDst[0] = pack2(f0.x, f0.y); aDst[1] = pack2(f0.z, f0.w);
      aDst[2] = pack2(f1.x, f1.y); aDst[3] = pack2(f1.z, f1.w);
      aDst[4] = pack2(f2.x, f2.y); aDst[5] = pack2(f2.z, f2.w);
      aDst[6] = pack2(f3.x, f3.y); aDst[7] = pack2(f3.z, f3.w);
      if (kb + 32 < Kdim) __builtin_prefetch((const char*)(src) + 128, 0, 1);
    }

    // ---- stage Bt tile (128 rows x 32 bf16) via Tensor Data Mover ----
    if (wid == 0) {
      unsigned long long ga =
          (unsigned long long)(const void*)(Bt + (size_t)tileN0 * Kdim + kb);
      v4i g0;
      g0[0] = 1;                                   // count=1, user descriptor
      g0[1] = (int)ldsB_base;                      // lds_addr
      g0[2] = (int)(unsigned)(ga & 0xFFFFFFFFu);   // global_addr[31:0]
      g0[3] = (int)(((unsigned)(ga >> 32) & 0x01FFFFFFu) |
                    0x80000000u);                  // global_addr[56:32] | type=2
      v8i g1;
      // data_size=1 (2B), pad_enable, pad_interval=3 (16 DW), pad_amount=3 (4 DW)
      g1[0] = (1 << 16) | (1 << 20) | (3 << 22) | (3 << 25);
      g1[1] = (int)(((unsigned)Kdim & 0xFFFFu) << 16);      // tensor_dim0 lo16
      g1[2] = (int)((((unsigned)Kdim >> 16) & 0xFFFFu) |    // tensor_dim0 hi16
                    (128u << 16));                          // tensor_dim1 lo16
      g1[3] = (int)(32u << 16);                             // tile_dim0 = 32
      g1[4] = 128;                                          // tile_dim1 = 128
      g1[5] = Kdim;                                         // dim0_stride lo32
      g1[6] = 0;                                            // stride hi / dim1_stride lo
      g1[7] = 0;
      asm volatile("tensor_load_to_lds %0, %1" ::"s"(g0), "s"(g1) : "memory");
      __builtin_amdgcn_s_wait_tensorcnt(0);
    }
    __syncthreads();

    // ---- build fragments per ISA 7.12.2 layouts ----
    v8u af[4], bf[2];
#pragma unroll
    for (int mt = 0; mt < 4; ++mt) {
      const unsigned int* ap = &ldsA[(waveM * 64 + mt * 16 + row16) * LSTR];
      uint4 g0 = *(const uint4*)(ap + halfSel * 4);        // K = half*8 + 0..7
      uint4 g1 = *(const uint4*)(ap + 8 + halfSel * 4);    // K = 16 + half*8 + 0..7
      v8u t;
      t[0] = g0.x; t[1] = g0.y; t[2] = g0.z; t[3] = g0.w;
      t[4] = g1.x; t[5] = g1.y; t[6] = g1.z; t[7] = g1.w;
      af[mt] = t;
    }
#pragma unroll
    for (int nt = 0; nt < 2; ++nt) {
      const unsigned int* bp =
          &ldsB[(waveN * 32 + nt * 16 + row16) * LSTR + halfSel * 4];
      uint4 g0 = *(const uint4*)(bp);
      uint4 g1 = *(const uint4*)(bp + 8);
      v8u t;
      t[0] = g0.x; t[1] = g0.y; t[2] = g0.z; t[3] = g0.w;
      t[4] = g1.x; t[5] = g1.y; t[6] = g1.z; t[7] = g1.w;
      bf[nt] = t;
    }
#pragma unroll
    for (int mt = 0; mt < 4; ++mt)
#pragma unroll
      for (int nt = 0; nt < 2; ++nt)
        acc[mt][nt] = __builtin_amdgcn_wmma_f32_16x16x32_bf16(
            false, __builtin_bit_cast(v16bf, af[mt]),
            false, __builtin_bit_cast(v16bf, bf[nt]),
            (short)0, acc[mt][nt], false, false);
    __syncthreads();
  }

  // ---- epilogue: D layout VGPR r -> M = r + (lane>>4)*8 ----
#pragma unroll
  for (int mt = 0; mt < 4; ++mt) {
#pragma unroll
    for (int nt = 0; nt < 2; ++nt) {
      int row0 = tileM0 + waveM * 64 + mt * 16 + halfSel * 8;
      int col  = tileN0 + waveN * 32 + nt * 16 + row16;
      float bv = bias ? bias[col] : 0.f;
#pragma unroll
      for (int r = 0; r < 8; ++r) {
        float v = acc[mt][nt][r] + bv;
        if (RELU) v = fmaxf(v, 0.f);
        size_t off = (size_t)(row0 + r) * (size_t)Ncols + col;
        if (OUT_BF16) ((unsigned short*)Cp)[off] = f2bf(v);
        else          ((float*)Cp)[off] = v;
      }
    }
  }
}

// ---------------------------------------------------------------------------
// Transpose + fp32->bf16 convert: W[K,N] -> Wt[N,K]
// ---------------------------------------------------------------------------
__global__ void transpose_bf16_kernel(const float* __restrict__ W,
                                      unsigned short* __restrict__ Wt,
                                      int K, int N) {
  int idx = blockIdx.x * 256 + threadIdx.x;
  if (idx >= K * N) return;
  int k = idx / N, n = idx - k * N;
  Wt[n * K + k] = f2bf(W[idx]);
}

// tanh(memory) (B,S,M) fp32 -> (S,B,M) bf16 (slot-major for per-slot GEMMs)
__global__ void tanh_prep_kernel(const float* __restrict__ memory,
                                 unsigned short* __restrict__ out) {
  int idx = blockIdx.x * 256 + threadIdx.x;  // < B*S*M
  int b = idx >> 11;
  int rem = idx & 2047;
  int s = rem >> 8;
  int m = rem & 255;
  out[((size_t)s * BB + b) * MM + m] = f2bf(tanhf(memory[idx]));
}

// mean_T(relu(rep_w * inputs)) : (B,T,M) bf16 -> (B,M) bf16
__global__ __launch_bounds__(256) void pool_kernel(
    const unsigned short* __restrict__ inputs, const float* __restrict__ rep_w,
    unsigned short* __restrict__ pooled) {
  int b = blockIdx.x;
  int m = threadIdx.x;
  float w = rep_w[m];
  float s = 0.f;
  const unsigned short* base = inputs + (size_t)b * TT * MM + m;
  for (int t = 0; t < TT; ++t) s += fmaxf(w * bf2f(base[t * MM]), 0.f);
  pooled[(size_t)b * MM + m] = f2bf(s * (1.f / (float)TT));
}

// ---------------------------------------------------------------------------
// Per-batch: attention over memory (q=mem(8), kv=inputs(64), topk=3),
// LN1, shared-weight MLP x2, LN2 -> mem2. One workgroup per batch, 256 thr.
// ---------------------------------------------------------------------------
__global__ __launch_bounds__(256) void attend_block_kernel(
    const float* __restrict__ memory, const unsigned short* __restrict__ k1,
    const unsigned short* __restrict__ v1, const float* __restrict__ Wq,
    const float* __restrict__ bq, const float* __restrict__ mlpW,
    const float* __restrict__ mlpb, const float* __restrict__ ln1s,
    const float* __restrict__ ln1b, const float* __restrict__ ln2s,
    const float* __restrict__ ln2b, float* __restrict__ mem2_out) {
  __shared__ float s_mem[SS * MM];      // original memory slots
  __shared__ float s_q[SS * MM];        // q proj, later mem1
  __shared__ float s_att[4 * SS * TT];  // attention (H,8,64), later h2
  __shared__ float s_a[SS * MM];        // att_out
  __shared__ float s_b[SS * MM];        // MLP hidden
  __shared__ unsigned short s_k[TT * MM];
  __shared__ unsigned short s_v[TT * MM];

  const int b = blockIdx.x;
  const int tid = threadIdx.x;
  const int lane = tid & 31;
  const int wv = tid >> 5;  // wave id == memory row for LN phases

  // ---- stage memory + k/v tiles ----
  const float* memb = memory + (size_t)b * SS * MM;
  for (int i = tid; i < SS * MM; i += 256) s_mem[i] = memb[i];
  {
    const uint4* ks = (const uint4*)(k1 + (size_t)b * TT * MM);
    const uint4* vs = (const uint4*)(v1 + (size_t)b * TT * MM);
    uint4* kd = (uint4*)s_k;
    uint4* vd = (uint4*)s_v;
    for (int i = tid; i < (TT * MM) / 8; i += 256) { kd[i] = ks[i]; vd[i] = vs[i]; }
  }
  __syncthreads();

  // ---- q = (mem @ Wq + bq) / sqrt(K) ---- thread: row=tid>>5, 8 cols
  {
    int r = tid >> 5;
    int c0 = (tid & 31) * 8;
    float acc[8];
#pragma unroll
    for (int j = 0; j < 8; ++j) acc[j] = bq[c0 + j];
    for (int k = 0; k < MM; ++k) {
      float a = s_mem[r * MM + k];
      const float* wr = Wq + (size_t)k * MM + c0;
      float4 w0 = *(const float4*)wr;
      float4 w1 = *(const float4*)(wr + 4);
      acc[0] += a * w0.x; acc[1] += a * w0.y; acc[2] += a * w0.z; acc[3] += a * w0.w;
      acc[4] += a * w1.x; acc[5] += a * w1.y; acc[6] += a * w1.z; acc[7] += a * w1.w;
    }
#pragma unroll
    for (int j = 0; j < 8; ++j) s_q[r * MM + c0 + j] = acc[j] * 0.125f;
  }
  __syncthreads();

  // ---- att[h][rq][m] = q . k ---- thread group of 8 per (h,rq)
  {
    int p = tid >> 3;          // 0..31 -> h=p>>3, rq=p&7
    int h = p >> 3, rq = p & 7;
    int m0 = (tid & 7) * 8;
    const float* qrow = &s_q[rq * MM + h * 64];
    for (int mm = 0; mm < 8; ++mm) {
      int m = m0 + mm;
      const unsigned short* krow = &s_k[m * MM + h * 64];
      float s = 0.f;
      for (int k = 0; k < 64; ++k) s += qrow[k] * bf2f(krow[k]);
      s_att[p * 64 + m] = s;
    }
  }
  __syncthreads();

  // ---- softmax + top-3 mask, one thread per (h,rq) row of 64 ----
  if (tid < 32) {
    float* row = &s_att[tid * 64];
    float mx = row[0];
    for (int i = 1; i < 64; ++i) mx = fmaxf(mx, row[i]);
    float sum = 0.f;
    for (int i = 0; i < 64; ++i) { float e = __expf(row[i] - mx); row[i] = e; sum += e; }
    float inv = 1.f / sum;
    for (int i = 0; i < 64; ++i) row[i] *= inv;
    float m1 = -1.f, m2 = -1.f, m3 = -1.f;
    for (int i = 0; i < 64; ++i) {
      float v = row[i];
      if (v > m1)      { m3 = m2; m2 = m1; m1 = v; }
      else if (v > m2) { m3 = m2; m2 = v; }
      else if (v > m3) { m3 = v; }
    }
    for (int i = 0; i < 64; ++i) row[i] = (row[i] >= m3) ? row[i] : 0.f;
  }
  __syncthreads();

  // ---- att_out = att @ v ----
  {
    int r = tid >> 5;
    int c0 = (tid & 31) * 8;
    int h = c0 >> 6;
    float acc[8] = {0.f, 0.f, 0.f, 0.f, 0.f, 0.f, 0.f, 0.f};
    const float* arow = &s_att[(h * 8 + r) * 64];
    for (int m = 0; m < 64; ++m) {
      float a = arow[m];
      const unsigned short* vrow = &s_v[m * MM + c0];
#pragma unroll
      for (int j = 0; j < 8; ++j) acc[j] += a * bf2f(vrow[j]);
    }
#pragma unroll
    for (int j = 0; j < 8; ++j) s_a[r * MM + c0 + j] = acc[j];
  }
  __syncthreads();

  // ---- mem1 = LN1(mem + att_out) ; wave per row, lane handles 8 elems ----
  {
    float vals[8];
    float sum = 0.f, sq = 0.f;
#pragma unroll
    for (int j = 0; j < 8; ++j) {
      int c = lane * 8 + j;
      float x = s_mem[wv * MM + c] + s_a[wv * MM + c];
      vals[j] = x; sum += x; sq += x * x;
    }
    for (int off = 16; off >= 1; off >>= 1) {
      sum += __shfl_xor(sum, off, 32);
      sq  += __shfl_xor(sq,  off, 32);
    }
    float mean = sum * (1.f / (float)MM);
    float var  = sq * (1.f / (float)MM) - mean * mean;
    float inv  = rsqrtf(var + 1e-5f);
#pragma unroll
    for (int j = 0; j < 8; ++j) {
      int c = lane * 8 + j;
      s_q[wv * MM + c] = (vals[j] - mean) * inv * ln1s[c] + ln1b[c];
    }
  }
  __syncthreads();

  // ---- MLP pass 1: h = relu(mem1 @ W + b) -> s_b ----
  {
    int r = tid >> 5;
    int c0 = (tid & 31) * 8;
    float acc[8];
#pragma unroll
    for (int j = 0; j < 8; ++j) acc[j] = mlpb[c0 + j];
    for (int k = 0; k < MM; ++k) {
      float a = s_q[r * MM + k];
      const float* wr = mlpW + (size_t)k * MM + c0;
      float4 w0 = *(const float4*)wr;
      float4 w1 = *(const float4*)(wr + 4);
      acc[0] += a * w0.x; acc[1] += a * w0.y; acc[2] += a * w0.z; acc[3] += a * w0.w;
      acc[4] += a * w1.x; acc[5] += a * w1.y; acc[6] += a * w1.z; acc[7] += a * w1.w;
    }
#pragma unroll
    for (int j = 0; j < 8; ++j) s_b[r * MM + c0 + j] = fmaxf(acc[j], 0.f);
  }
  __syncthreads();

  // ---- MLP pass 2 (shared weights): h2 = relu(h @ W + b) -> s_att ----
  {
    int r = tid >> 5;
    int c0 = (tid & 31) * 8;
    float acc[8];
#pragma unroll
    for (int j = 0; j < 8; ++j) acc[j] = mlpb[c0 + j];
    for (int k = 0; k < MM; ++k) {
      float a = s_b[r * MM + k];
      const float* wr = mlpW + (size_t)k * MM + c0;
      float4 w0 = *(const float4*)wr;
      float4 w1 = *(const float4*)(wr + 4);
      acc[0] += a * w0.x; acc[1] += a * w0.y; acc[2] += a * w0.z; acc[3] += a * w0.w;
      acc[4] += a * w1.x; acc[5] += a * w1.y; acc[6] += a * w1.z; acc[7] += a * w1.w;
    }
#pragma unroll
    for (int j = 0; j < 8; ++j) s_att[r * MM + c0 + j] = fmaxf(acc[j], 0.f);
  }
  __syncthreads();

  // ---- mem2 = LN2(mem1 + h2) -> global ----
  {
    float vals[8];
    float sum = 0.f, sq = 0.f;
#pragma unroll
    for (int j = 0; j < 8; ++j) {
      int c = lane * 8 + j;
      float x = s_q[wv * MM + c] + s_att[wv * MM + c];
      vals[j] = x; sum += x; sq += x * x;
    }
    for (int off = 16; off >= 1; off >>= 1) {
      sum += __shfl_xor(sum, off, 32);
      sq  += __shfl_xor(sq,  off, 32);
    }
    float mean = sum * (1.f / (float)MM);
    float var  = sq * (1.f / (float)MM) - mean * mean;
    float inv  = rsqrtf(var + 1e-5f);
    float* dst = mem2_out + (size_t)b * SS * MM;
#pragma unroll
    for (int j = 0; j < 8; ++j) {
      int c = lane * 8 + j;
      dst[wv * MM + c] = (vals[j] - mean) * inv * ln2s[c] + ln2b[c];
    }
  }
}

// ---------------------------------------------------------------------------
// Gate combine: next_memory = sig(gm+gi+ib)*tanh(mem2) + sig(...+fb)*memory.
// Writes logits region, next_memory region, and bf16 copy for k2/v2 GEMMs.
// ---------------------------------------------------------------------------
__global__ void combine_kernel(const float* __restrict__ memory,
                               const float* __restrict__ mem2,
                               const float* __restrict__ gate_mem,  // (S,B,G)
                               const float* __restrict__ gate_in,   // (B,G)
                               const float* __restrict__ input_bias,
                               const float* __restrict__ forget_bias,
                               float* __restrict__ out_logits,
                               float* __restrict__ out_nextmem,
                               unsigned short* __restrict__ next_bf16) {
  int idx = blockIdx.x * 256 + threadIdx.x;  // < B*S*M
  int b = idx >> 11;
  int rem = idx & 2047;
  int s = rem >> 8;
  int m = rem & 255;
  size_t gm = ((size_t)s * BB + b) * GG;
  size_t gi = (size_t)b * GG;
  float xg  = gate_mem[gm + m]      + gate_in[gi + m]      + input_bias[0];
  float fgv = gate_mem[gm + MM + m] + gate_in[gi + MM + m] + forget_bias[0];
  float ig = sigmoidf(xg);
  float fg = sigmoidf(fgv);
  float nm = ig * tanhf(mem2[idx]) + fg * memory[idx];
  out_logits[idx]  = nm;
  out_nextmem[idx] = nm;
  next_bf16[idx]   = f2bf(nm);
}

// ---------------------------------------------------------------------------
// hx: q = inputs-proj (B,T,256), kv = next_memory (8 slots), softmax, no topk.
// One workgroup per batch; thread = (t, h).
// ---------------------------------------------------------------------------
__global__ __launch_bounds__(256) void hx_attend_kernel(
    const unsigned short* __restrict__ qhx, const unsigned short* __restrict__ k2,
    const unsigned short* __restrict__ v2, float* __restrict__ hx_out) {
  __shared__ float s_kf[SS * MM];
  __shared__ float s_vf[SS * MM];
  const int b = blockIdx.x;
  const int tid = threadIdx.x;
  const unsigned short* kb_ = k2 + (size_t)b * SS * MM;
  const unsigned short* vb_ = v2 + (size_t)b * SS * MM;
  for (int i = tid; i < SS * MM; i += 256) { s_kf[i] = bf2f(kb_[i]); s_vf[i] = bf2f(vb_[i]); }
  __syncthreads();

  const int t = tid >> 2;
  const int h = tid & 3;
  const unsigned short* qrow = qhx + ((size_t)b * TT + t) * MM + h * 64;
  float q[64];
#pragma unroll
  for (int i = 0; i < 64; ++i) q[i] = bf2f(qrow[i]);

  float a[SS];
  float mx = -3.4e38f;
#pragma unroll
  for (int m = 0; m < SS; ++m) {
    const float* kr = &s_kf[m * MM + h * 64];
    float s = 0.f;
    for (int k = 0; k < 64; ++k) s += q[k] * kr[k];
    s *= 0.125f;
    a[m] = s;
    mx = fmaxf(mx, s);
  }
  float sum = 0.f;
#pragma unroll
  for (int m = 0; m < SS; ++m) { a[m] = __expf(a[m] - mx); sum += a[m]; }
  float inv = 1.f / sum;
#pragma unroll
  for (int m = 0; m < SS; ++m) a[m] *= inv;

  float* orow = hx_out + ((size_t)b * TT + t) * MM + h * 64;
  for (int k = 0; k < 64; ++k) {
    float o = 0.f;
#pragma unroll
    for (int m = 0; m < SS; ++m) o += a[m] * s_vf[m * MM + h * 64 + k];
    orow[k] = o;
  }
}

// ---------------------------------------------------------------------------
// Host-side orchestration.
// ---------------------------------------------------------------------------
extern "C" void kernel_launch(void* const* d_in, const int* in_sizes, int n_in,
                              void* d_out, int out_size, void* d_ws,
                              size_t ws_size, hipStream_t stream) {
  (void)in_sizes; (void)n_in; (void)out_size; (void)ws_size;

  const float* ipts   = (const float*)d_in[0];
  const float* memory = (const float*)d_in[1];
  const float* Wq     = (const float*)d_in[2];
  const float* bq     = (const float*)d_in[3];
  const float* Wk     = (const float*)d_in[4];
  const float* bk     = (const float*)d_in[5];
  const float* Wv     = (const float*)d_in[6];
  const float* bv     = (const float*)d_in[7];
  const float* mlpW   = (const float*)d_in[8];
  const float* mlpb   = (const float*)d_in[9];
  const float* ln1s   = (const float*)d_in[10];
  const float* ln1b   = (const float*)d_in[11];
  const float* ln2s   = (const float*)d_in[12];
  const float* ln2b   = (const float*)d_in[13];
  const float* inW    = (const float*)d_in[14];
  const float* inb    = (const float*)d_in[15];
  const float* repw   = (const float*)d_in[16];
  const float* ginW   = (const float*)d_in[17];
  const float* ginb   = (const float*)d_in[18];
  const float* gmemW  = (const float*)d_in[19];
  const float* gmemb  = (const float*)d_in[20];
  const float* fbias  = (const float*)d_in[21];
  const float* ibias  = (const float*)d_in[22];

  float* out = (float*)d_out;
  float* out_logits  = out;               // B*S*M
  float* out_nextmem = out + BSM;         // B*S*M
  float* out_hx      = out + 2 * BSM;     // B*T*M

  // ---- workspace carve-up (256B aligned regions) ----
  char* ws = (char*)d_ws;
  size_t off = 0;
  auto carve = [&](size_t bytes) -> void* {
    off = (off + 255) & ~(size_t)255;
    void* p = ws + off;
    off += bytes;
    return p;
  };
  unsigned short* inputs_bf = (unsigned short*)carve((size_t)BT * MM * 2);
  unsigned short* k1        = (unsigned short*)carve((size_t)BT * MM * 2);
  unsigned short* v1        = (unsigned short*)carve((size_t)BT * MM * 2);
  unsigned short* qhx       = (unsigned short*)carve((size_t)BT * MM * 2);
  unsigned short* Wt_in     = (unsigned short*)carve((size_t)MM * MM * 2);
  unsigned short* Wt_q      = (unsigned short*)carve((size_t)MM * MM * 2);
  unsigned short* Wt_k      = (unsigned short*)carve((size_t)MM * MM * 2);
  unsigned short* Wt_v      = (unsigned short*)carve((size_t)MM * MM * 2);
  unsigned short* Wt_gin    = (unsigned short*)carve((size_t)GG * MM * 2);
  unsigned short* Wt_gmem   = (unsigned short*)carve((size_t)SS * GG * MM * 2);
  unsigned short* pooled    = (unsigned short*)carve((size_t)BB * MM * 2);
  float*          gate_in   = (float*)carve((size_t)BB * GG * 4);
  float*          gate_mem  = (float*)carve((size_t)SS * BB * GG * 4);
  unsigned short* tanhm     = (unsigned short*)carve((size_t)SS * BB * MM * 2);
  float*          mem2      = (float*)carve((size_t)BB * SS * MM * 4);
  unsigned short* nextbf    = (unsigned short*)carve((size_t)BB * SS * MM * 2);
  unsigned short* k2        = (unsigned short*)carve((size_t)BB * SS * MM * 2);
  unsigned short* v2        = (unsigned short*)carve((size_t)BB * SS * MM * 2);

  // ---- 1. weight transposes (fp32 -> bf16, (K,N)->(N,K)) ----
  int n256 = (MM * MM + 255) / 256;
  transpose_bf16_kernel<<<n256, 256, 0, stream>>>(inW, Wt_in, MM, MM);
  transpose_bf16_kernel<<<n256, 256, 0, stream>>>(Wq, Wt_q, MM, MM);
  transpose_bf16_kernel<<<n256, 256, 0, stream>>>(Wk, Wt_k, MM, MM);
  transpose_bf16_kernel<<<n256, 256, 0, stream>>>(Wv, Wt_v, MM, MM);
  transpose_bf16_kernel<<<(MM * GG + 255) / 256, 256, 0, stream>>>(ginW, Wt_gin, MM, GG);
  for (int s = 0; s < SS; ++s)
    transpose_bf16_kernel<<<(MM * GG + 255) / 256, 256, 0, stream>>>(
        gmemW + (size_t)s * MM * GG, Wt_gmem + (size_t)s * GG * MM, MM, GG);

  // ---- 2. inputs = ipts @ in_W + in_b (bf16 out) ----
  dim3 gBT(MM / 128, BT / 128);
  gemm_wmma_kernel<false, true, false><<<gBT, 256, 0, stream>>>(
      ipts, Wt_in, inb, inputs_bf, BT, MM, MM);

  // ---- 3. k1 / v1 / q_hx projections of inputs ----
  gemm_wmma_kernel<true, true, false><<<gBT, 256, 0, stream>>>(
      inputs_bf, Wt_k, bk, k1, BT, MM, MM);
  gemm_wmma_kernel<true, true, false><<<gBT, 256, 0, stream>>>(
      inputs_bf, Wt_v, bv, v1, BT, MM, MM);
  gemm_wmma_kernel<true, true, false><<<gBT, 256, 0, stream>>>(
      inputs_bf, Wt_q, bq, qhx, BT, MM, MM);

  // ---- 4. attend-over-memory block -> mem2 ----
  attend_block_kernel<<<BB, 256, 0, stream>>>(memory, k1, v1, Wq, bq, mlpW, mlpb,
                                              ln1s, ln1b, ln2s, ln2b, mem2);

  // ---- 5. gate_in = mean_T(relu(rep_w*inputs)) @ gin_W + gin_b ----
  pool_kernel<<<BB, 256, 0, stream>>>(inputs_bf, repw, pooled);
  dim3 gGate(GG / 128, BB / 128);
  gemm_wmma_kernel<true, false, false><<<gGate, 256, 0, stream>>>(
      pooled, Wt_gin, ginb, gate_in, BB, GG, MM);

  // ---- 6. gate_mem: per-slot weight-stationary GEMMs ----
  tanh_prep_kernel<<<BSM / 256, 256, 0, stream>>>(memory, tanhm);
  for (int s = 0; s < SS; ++s)
    gemm_wmma_kernel<true, false, false><<<gGate, 256, 0, stream>>>(
        tanhm + (size_t)s * BB * MM, Wt_gmem + (size_t)s * GG * MM,
        gmemb + (size_t)s * GG, gate_mem + (size_t)s * BB * GG, BB, GG, MM);

  // ---- 7. gates + next_memory (logits + next_memory outputs) ----
  combine_kernel<<<BSM / 256, 256, 0, stream>>>(memory, mem2, gate_mem, gate_in,
                                                ibias, fbias, out_logits,
                                                out_nextmem, nextbf);

  // ---- 8. k2 / v2 projections of next_memory ----
  dim3 gNM(MM / 128, (BB * SS) / 128);
  gemm_wmma_kernel<true, true, false><<<gNM, 256, 0, stream>>>(
      nextbf, Wt_k, bk, k2, BB * SS, MM, MM);
  gemm_wmma_kernel<true, true, false><<<gNM, 256, 0, stream>>>(
      nextbf, Wt_v, bv, v2, BB * SS, MM, MM);

  // ---- 9. hx attention -> hx output ----
  hx_attend_kernel<<<BB, 256, 0, stream>>>(qhx, k2, v2, out_hx);
}
